// SelfAttention_23553600651733
// MI455X (gfx1250) — matrix operand order repaired and checked by reading, not compiled, not run on hardware
//
#include <hip/hip_runtime.h>
#include <hip/hip_bf16.h>

typedef __attribute__((ext_vector_type(16))) _Float16 v16h;
typedef __attribute__((ext_vector_type(2)))  _Float16 v2h;
typedef __attribute__((ext_vector_type(8)))  float    v8f;

#define BATCH   8
#define CCH     256      // channels
#define NHEADS  8
#define HDIM    32       // head dim
#define NPIX    1024     // 32*32

// ---------------------------------------------------------------------------
// WMMA wrapper: D = A(16x32 f16) * B(32x16 f16) + C(16x16 f32)
// ---------------------------------------------------------------------------
__device__ __forceinline__ v8f wmma_f16(v16h a, v16h b, v8f c) {
    return __builtin_amdgcn_wmma_f32_16x16x32_f16(
        /*neg_a=*/false, a, /*neg_b=*/false, b,
        /*c_mod=*/(short)0, c, /*reuse_a=*/false, /*reuse_b=*/false);
}

// ---------------------------------------------------------------------------
// Operand loaders matching CDNA5 ISA 7.12.2 VGPR layouts (wave32).
// A (16x32 f16): lane L holds row M=L&15.
//   VGPR r<4 : K = {2r,2r+1}   (lanes 0-15) / {8+2r,9+2r}   (lanes 16-31)
//   VGPR r>=4: K = {16+2(r-4),...} (lanes 0-15) / {24+2(r-4),...} (16-31)
// All K pair bases are even -> pair-interleaved layouts load one u32 per pair.
// ---------------------------------------------------------------------------
__device__ __forceinline__ int a_kbase(int r, int hi) {
    return (r < 4) ? ((hi ? 8 : 0) + 2 * r) : ((hi ? 24 : 16) + 2 * (r - 4));
}

// generic (strided) A loader: elem(m,k) = src[(row0+m)*rs + (k0+k)*ks]
template <typename T>
__device__ __forceinline__ v16h load_a16(const T* __restrict__ src,
                                         int row0, long rs, int k0, long ks) {
    int lane = threadIdx.x & 31;
    int m = lane & 15;
    int hi = lane >> 4;
    v16h a;
#pragma unroll
    for (int r = 0; r < 8; ++r) {
        int kb = a_kbase(r, hi);
        const T* p = src + (long)(row0 + m) * rs + (long)(k0 + kb) * ks;
        a[2 * r]     = (_Float16)p[0];
        a[2 * r + 1] = (_Float16)p[ks];
    }
    return a;
}

// A loader from pair-interleaved [c-pairs][NPIX] layout:
// elem(m,k) = pk[((k0+k)>>1)*NPIX + row0+m][ (k0+k)&1 ]   (k0 even)
__device__ __forceinline__ v16h load_a16_pk(const v2h* __restrict__ pk,
                                            int row0, int k0) {
    int lane = threadIdx.x & 31;
    int m = lane & 15;
    int hi = lane >> 4;
    v16h a;
#pragma unroll
    for (int r = 0; r < 8; ++r) {
        int kb = a_kbase(r, hi);
        v2h p = pk[(long)((k0 + kb) >> 1) * NPIX + row0 + m];
        a[2 * r]     = p[0];
        a[2 * r + 1] = p[1];
    }
    return a;
}

// generic (strided) B loader (32x16): lane L holds column N=L&15.
//   VGPR r: K = a_kbase(r, hi) + {0,1} (same K order as A)
// elem(k,n) = src[(k0+k)*ks + (col0+n)*cs]
template <typename T>
__device__ __forceinline__ v16h load_b16(const T* __restrict__ src,
                                         int k0, long ks, int col0, long cs) {
    int lane = threadIdx.x & 31;
    int n = lane & 15;
    int hi = lane >> 4;
    v16h b;
#pragma unroll
    for (int r = 0; r < 8; ++r) {
        int kb = a_kbase(r, hi);
        const T* p = src + (long)(k0 + kb) * ks + (long)(col0 + n) * cs;
        b[2 * r]     = (_Float16)p[0];
        b[2 * r + 1] = (_Float16)p[ks];
    }
    return b;
}

// B loader from pair-interleaved [c-pairs][NPIX] layout (k0 even)
__device__ __forceinline__ v16h load_b16_pk(const v2h* __restrict__ pk,
                                            int k0, int col0) {
    int lane = threadIdx.x & 31;
    int n = lane & 15;
    int hi = lane >> 4;
    v16h b;
#pragma unroll
    for (int r = 0; r < 8; ++r) {
        int kb = a_kbase(r, hi);
        v2h p = pk[(long)((k0 + kb) >> 1) * NPIX + col0 + n];
        b[2 * r]     = p[0];
        b[2 * r + 1] = p[1];
    }
    return b;
}

// ---------------------------------------------------------------------------
// Kernel 0a: f32 -> f16 plain conversion (weights)
// ---------------------------------------------------------------------------
__global__ void cvt_f16_kernel(const float* __restrict__ src,
                               _Float16* __restrict__ dst, int count) {
    int i = blockIdx.x * blockDim.x + threadIdx.x;
    if (i < count) dst[i] = (_Float16)src[i];
}

// Kernel 0b: f32 -> f16 with channel-pair interleave: (b,c,n) ->
//   base_b + ((c>>1)*NPIX + n)*2 + (c&1)
__global__ void cvt_pack_x_kernel(const float* __restrict__ src,
                                  _Float16* __restrict__ dst) {
    int i = blockIdx.x * blockDim.x + threadIdx.x;   // 0 .. BATCH*CCH*NPIX-1
    int n   = i & (NPIX - 1);
    int c   = (i >> 10) & (CCH - 1);
    int b   = i >> 18;
    long o  = (long)b * CCH * NPIX + ((long)(c >> 1) * NPIX + n) * 2 + (c & 1);
    dst[o] = (_Float16)src[i];
}

// ---------------------------------------------------------------------------
// Kernel 1: QKV projections. Y[o,n] = sum_c W[o,c] * x[b,c,n] + bias[o]
// A = f16 weights (plain, contiguous-K -> b128 merges); B = pair-packed x.
// Q/K outputs stored pair-interleaved (for K2's strided reads); V plain.
// Q is pre-scaled by 1/sqrt(HDIM). 64x16 slab per wave (4 WMMAs / B tile).
// grid (64, 3, 8), block 128 (4 waves)
// ---------------------------------------------------------------------------
__global__ void qkv_proj_kernel(const _Float16* __restrict__ xpk,
                                const _Float16* __restrict__ Wq16,
                                const _Float16* __restrict__ Wk16,
                                const _Float16* __restrict__ Wv16,
                                const float* __restrict__ bq,
                                const float* __restrict__ bk,
                                const float* __restrict__ bv,
                                _Float16* __restrict__ q_ws,   // packed
                                _Float16* __restrict__ k_ws,   // packed
                                _Float16* __restrict__ v_ws) { // plain
    int wave = threadIdx.x >> 5;
    int task = blockIdx.x * 4 + wave;          // 4 o-groups * 64 n-tiles = 256
    int p    = blockIdx.y;                     // 0=q 1=k 2=v
    int b    = blockIdx.z;
    int o0 = (task >> 6) * 64;                 // 64-row M slab
    int n0 = (task & 63) * 16;

    const _Float16* W = (p == 0) ? Wq16 : (p == 1) ? Wk16 : Wv16;
    const float* bias = (p == 0) ? bq : (p == 1) ? bk : bv;
    const v2h* xb = (const v2h*)(xpk + (long)b * CCH * NPIX);

    v8f acc[4] = {v8f{}, v8f{}, v8f{}, v8f{}};
    for (int k0 = 0; k0 < CCH; k0 += 32) {
        v16h bm = load_b16_pk(xb, k0, n0);
#pragma unroll
        for (int t = 0; t < 4; ++t) {
            v16h a = load_a16<_Float16>(W, o0 + 16 * t, CCH, k0, 1);
            acc[t] = wmma_f16(a, bm, acc[t]);
        }
    }

    int lane = threadIdx.x & 31;
    int n = lane & 15, hi = lane >> 4;
    float scale = (p == 0) ? 0.17677669529663689f : 1.0f;  // 1/sqrt(32)
    _Float16* db = ((p == 0) ? q_ws : (p == 1) ? k_ws : v_ws)
                   + (long)b * CCH * NPIX;
#pragma unroll
    for (int t = 0; t < 4; ++t) {
#pragma unroll
        for (int r = 0; r < 8; ++r) {
            int row = o0 + 16 * t + r + (hi ? 8 : 0);
            float v = (acc[t][r] + bias[row]) * scale;
            long idx = (p == 2)
                ? (long)row * NPIX + n0 + n
                : ((long)(row >> 1) * NPIX + n0 + n) * 2 + (row & 1);
            db[idx] = (_Float16)v;
        }
    }
}

// ---------------------------------------------------------------------------
// Kernel 2: flash-style attention per (b, h, 16-row query tile).
// S = (scaled Q)^T K via WMMA; two-pass softmax (row-max, then exp/sum);
// P staged through LDS (C-layout -> A-layout transpose); O += P * V^T.
// Q/K read pair-packed (u32 per K-pair); V read plain (K contiguous).
// grid (16, 8, 8), block 128 (4 waves), LDS 4KB
// ---------------------------------------------------------------------------
__global__ void attn_kernel(const _Float16* __restrict__ q_ws,  // packed
                            const _Float16* __restrict__ k_ws,  // packed
                            const _Float16* __restrict__ v_ws,  // plain
                            _Float16* __restrict__ o_ws) {      // packed out
    __shared__ _Float16 lds_p[4][16 * 32];

    int wave = threadIdx.x >> 5;
    int lane = threadIdx.x & 31;
    int ib = blockIdx.x * 4 + wave;            // query tile 0..63
    int h  = blockIdx.y;
    int b  = blockIdx.z;
    int i0 = ib * 16;

    // head base in pair units: channel c0 = h*HDIM (even)
    const v2h* Qp = (const v2h*)q_ws + ((long)b * (CCH / 2) + h * (HDIM / 2)) * NPIX;
    const v2h* Kp = (const v2h*)k_ws + ((long)b * (CCH / 2) + h * (HDIM / 2)) * NPIX;
    const _Float16* V = v_ws + ((long)b * CCH + h * HDIM) * NPIX;

    // A = Q^T tile: elem(m, k=d) = Q[d][i0+m]
    v16h aq = load_a16_pk(Qp, i0, 0);

    // ---- pass 1: per-row max of S over all j ----
    float rm[8];
#pragma unroll
    for (int r = 0; r < 8; ++r) rm[r] = -1e30f;
    for (int j0 = 0; j0 < NPIX; j0 += 16) {
        v16h bk = load_b16_pk(Kp, 0, j0);
        v8f c = {};
        c = wmma_f16(aq, bk, c);
#pragma unroll
        for (int r = 0; r < 8; ++r) rm[r] = fmaxf(rm[r], c[r]);
    }
    // reduce across the 16-lane half (masks < 16 stay within each half)
#pragma unroll
    for (int r = 0; r < 8; ++r) {
#pragma unroll
        for (int m = 1; m < 16; m <<= 1)
            rm[r] = fmaxf(rm[r], __shfl_xor(rm[r], m, 32));
    }

    // ---- pass 2: recompute S, exponentiate, accumulate O = P * V^T ----
    float ls[8];
#pragma unroll
    for (int r = 0; r < 8; ++r) ls[r] = 0.0f;
    v8f oacc0 = {}, oacc1 = {};
    _Float16* pt = lds_p[wave];
    int n = lane & 15, hi = lane >> 4;

    for (int j0 = 0; j0 < NPIX; j0 += 32) {
#pragma unroll
        for (int t = 0; t < 2; ++t) {
            v16h bk = load_b16_pk(Kp, 0, j0 + 16 * t);
            v8f c = {};
            c = wmma_f16(aq, bk, c);
#pragma unroll
            for (int r = 0; r < 8; ++r) {
                float e = __expf(c[r] - rm[r]);
                ls[r] += e;
                int row = r + (hi ? 8 : 0);
                pt[row * 32 + 16 * t + n] = (_Float16)e;   // P tile 16x32 in LDS
            }
        }
        __syncthreads();
        // A = P (16 rows i x 32 cols j) from LDS (contiguous K -> b32 merges)
        v16h ap = load_a16<_Float16>(pt, 0, 32, 0, 1);
        // B = V^T chunk: elem(kk, d') = V[d'*NPIX + j0+kk]  (K contiguous)
        v16h bv0 = load_b16<_Float16>(V, j0, 1, 0, NPIX);
        v16h bv1 = load_b16<_Float16>(V, j0, 1, 16, NPIX);
        oacc0 = wmma_f16(ap, bv0, oacc0);
        oacc1 = wmma_f16(ap, bv1, oacc1);
        __syncthreads();
    }

    // reduce row sums across half-lanes
#pragma unroll
    for (int r = 0; r < 8; ++r) {
#pragma unroll
        for (int m = 1; m < 16; m <<= 1)
            ls[r] += __shfl_xor(ls[r], m, 32);
    }

    // store O^T pair-packed as channel c = h*HDIM + d', position i
    _Float16* Ob = o_ws + (long)b * CCH * NPIX;
#pragma unroll
    for (int r = 0; r < 8; ++r) {
        int row_i = i0 + r + (hi ? 8 : 0);
        float inv = 1.0f / ls[r];
        int c0 = h * HDIM + n;
        int c1 = h * HDIM + 16 + n;
        Ob[((long)(c0 >> 1) * NPIX + row_i) * 2 + (c0 & 1)] =
            (_Float16)(oacc0[r] * inv);
        Ob[((long)(c1 >> 1) * NPIX + row_i) * 2 + (c1 & 1)] =
            (_Float16)(oacc1[r] * inv);
    }
}

// ---------------------------------------------------------------------------
// Kernel 3: output projection + bias + residual, f32 result.
// A = f16 Wo (plain); B = pair-packed attention output. 64x16 slab per wave.
// grid (64, 8), block 128
// ---------------------------------------------------------------------------
__global__ void out_proj_kernel(const float* __restrict__ x,
                                const _Float16* __restrict__ Wo16,
                                const float* __restrict__ bo,
                                const _Float16* __restrict__ o_ws, // packed
                                float* __restrict__ out) {
    int wave = threadIdx.x >> 5;
    int task = blockIdx.x * 4 + wave;
    int b    = blockIdx.y;
    int o0 = (task >> 6) * 64;
    int n0 = (task & 63) * 16;

    const v2h* Ob = (const v2h*)(o_ws + (long)b * CCH * NPIX);
    v8f acc[4] = {v8f{}, v8f{}, v8f{}, v8f{}};
    for (int k0 = 0; k0 < CCH; k0 += 32) {
        v16h bm = load_b16_pk(Ob, k0, n0);
#pragma unroll
        for (int t = 0; t < 4; ++t) {
            v16h a = load_a16<_Float16>(Wo16, o0 + 16 * t, CCH, k0, 1);
            acc[t] = wmma_f16(a, bm, acc[t]);
        }
    }

    int lane = threadIdx.x & 31;
    int n = lane & 15, hi = lane >> 4;
    const float* xb = x + (long)b * CCH * NPIX;
    float* ob = out + (long)b * CCH * NPIX;
#pragma unroll
    for (int t = 0; t < 4; ++t) {
#pragma unroll
        for (int r = 0; r < 8; ++r) {
            int row = o0 + 16 * t + r + (hi ? 8 : 0);
            long idx = (long)row * NPIX + n0 + n;
            ob[idx] = acc[t][r] + bo[row] + xb[idx];
        }
    }
}

// ---------------------------------------------------------------------------
// Kernel 4: attn_map. Mean over j of a softmax over j is exactly 1/N;
// head-mean of distributions is a distribution, so attn_map == 1/1024.
// ---------------------------------------------------------------------------
__global__ void attn_map_kernel(float* __restrict__ dst, int count, float val) {
    int i = blockIdx.x * blockDim.x + threadIdx.x;
    if (i < count) dst[i] = val;
}

// ---------------------------------------------------------------------------
extern "C" void kernel_launch(void* const* d_in, const int* in_sizes, int n_in,
                              void* d_out, int out_size, void* d_ws, size_t ws_size,
                              hipStream_t stream) {
    const float* x  = (const float*)d_in[0];
    const float* Wq = (const float*)d_in[1];
    const float* bq = (const float*)d_in[2];
    const float* Wk = (const float*)d_in[3];
    const float* bk = (const float*)d_in[4];
    const float* Wv = (const float*)d_in[5];
    const float* bv = (const float*)d_in[6];
    const float* Wo = (const float*)d_in[7];
    const float* bo = (const float*)d_in[8];

    float* out = (float*)d_out;

    const size_t per = (size_t)BATCH * CCH * NPIX;   // 2M f16 elements each
    const size_t wsz = (size_t)CCH * CCH;            // 65536 per weight
    _Float16* xpk  = (_Float16*)d_ws;
    _Float16* q_ws = xpk + per;
    _Float16* k_ws = q_ws + per;
    _Float16* v_ws = k_ws + per;
    _Float16* o_ws = v_ws + per;
    _Float16* Wq16 = o_ws + per;
    _Float16* Wk16 = Wq16 + wsz;
    _Float16* Wv16 = Wk16 + wsz;
    _Float16* Wo16 = Wv16 + wsz;                     // ~20.5 MiB total

    const int xcount = (int)per;                     // 2,097,152
    cvt_pack_x_kernel<<<xcount / 256, 256, 0, stream>>>(x, xpk);
    cvt_f16_kernel<<<(int)wsz / 256, 256, 0, stream>>>(Wq, Wq16, (int)wsz);
    cvt_f16_kernel<<<(int)wsz / 256, 256, 0, stream>>>(Wk, Wk16, (int)wsz);
    cvt_f16_kernel<<<(int)wsz / 256, 256, 0, stream>>>(Wv, Wv16, (int)wsz);
    cvt_f16_kernel<<<(int)wsz / 256, 256, 0, stream>>>(Wo, Wo16, (int)wsz);

    qkv_proj_kernel<<<dim3(64, 3, BATCH), 128, 0, stream>>>(
        xpk, Wq16, Wk16, Wv16, bq, bk, bv, q_ws, k_ws, v_ws);

    attn_kernel<<<dim3(16, NHEADS, BATCH), 128, 0, stream>>>(
        q_ws, k_ws, v_ws, o_ws);

    out_proj_kernel<<<dim3(64, BATCH), 128, 0, stream>>>(
        x, Wo16, bo, o_ws, out);

    const int map_elems = BATCH * 32 * 32;           // 8192
    attn_map_kernel<<<(map_elems + 255) / 256, 256, 0, stream>>>(
        out + per, map_elems, 1.0f / (float)NPIX);
}
